// Head_2482491097827
// MI455X (gfx1250) — matrix operand order repaired and checked by reading, not compiled, not run on hardware
//
#include <hip/hip_runtime.h>
#include <hip/hip_bf16.h>
#include <math.h>

// ---------------------------------------------------------------------------
// Causal single-head attention, CDNA5 (gfx1250, wave32) WMMA implementation.
//   B=4, T=2048, D=1024, HS=64.  bf16 matmuls, f32 accumulate, f32 softmax.
//   k0: W transpose/convert -> bf16 [3][HS][D]
//   k1: projection, one wave per (16-row tile, matrix), software
//       double-buffered A/B tiles so WMMAs never wait on just-issued loads.
//   k2: flash attention, one wave per (batch, 16-query tile), K/V tiles
//       prefetched one key-block ahead (latency hidden under softmax VALU).
// ---------------------------------------------------------------------------

typedef __attribute__((ext_vector_type(16))) __bf16 v16bf;
typedef __attribute__((ext_vector_type(8)))  __bf16 v8bf;
typedef __attribute__((ext_vector_type(8)))  float  v8f;

#define BATCH 4
#define TSEQ  2048
#define DMODEL 1024
#define HSIZE 64

static __device__ __forceinline__ v8f wmma_bf16(v16bf a, v16bf b, v8f c) {
    return __builtin_amdgcn_wmma_f32_16x16x32_bf16(
        false, a, false, b, (short)0, c, false, false);
}

// 16 contiguous bf16 (32 bytes, 16B-aligned) as two 128-bit loads.
static __device__ __forceinline__ v16bf load16bf(const __bf16* p) {
    const v8bf* p8 = (const v8bf*)p;
    v8bf lo = p8[0], hi = p8[1];
    return __builtin_shufflevector(lo, hi, 0, 1, 2, 3, 4, 5, 6, 7,
                                           8, 9, 10, 11, 12, 13, 14, 15);
}

// ---------------------------------------------------------------------------
// Kernel 0: W transpose+convert.  wt[mm][h][k] = (bf16) W_mm[k][h].
// ---------------------------------------------------------------------------
__global__ __launch_bounds__(256)
void wtrans_kernel(const float* __restrict__ Wq,
                   const float* __restrict__ Wk,
                   const float* __restrict__ Wv,
                   __bf16* __restrict__ wt)      // [3][HS][D]
{
    const int mm = blockIdx.y;
    const float* W = (mm == 0) ? Wq : (mm == 1) ? Wk : Wv;
    const int h     = threadIdx.x & 63;
    const int kbase = blockIdx.x * 64 + (threadIdx.x >> 6) * 16;
    #pragma unroll
    for (int i = 0; i < 16; ++i) {
        const int kk = kbase + i;
        wt[((size_t)mm * HSIZE + h) * DMODEL + kk] = (__bf16)W[(size_t)kk * HSIZE + h];
    }
}

// ---------------------------------------------------------------------------
// Kernel 1: projection for ONE matrix per wave, double-buffered K-loop.
//   blockIdx.x = 16-row tile (512), blockIdx.y = matrix (0=q, 1=k, 2=v).
// ---------------------------------------------------------------------------
__global__ __launch_bounds__(32)
void proj_kernel(const float* __restrict__ x,
                 const __bf16* __restrict__ wt,  // [3][HS][D]
                 __bf16* __restrict__ qb,        // [B*T, 64]
                 __bf16* __restrict__ kb,        // [B*T, 64]
                 __bf16* __restrict__ vtb)       // [B][64][T]
{
    const int lane = threadIdx.x & 31;
    const int m    = lane & 15;
    const int half = lane >> 4;
    const int row0 = blockIdx.x * 16;
    const int mm   = blockIdx.y;

    const __bf16* wbase = wt + (size_t)mm * HSIZE * DMODEL;
    const float*  xrow  = x + (size_t)(row0 + m) * DMODEL + half * 16;

#define LOAD_A(K0, G0, G1, G2, G3) do {                                  \
        const float4* _xa = (const float4*)(xrow + (K0));                \
        const float4* _xb = (const float4*)(xrow + (K0) + 8);            \
        G0 = _xa[0]; G1 = _xa[1]; G2 = _xb[0]; G3 = _xb[1]; } while (0)

#define CVT_A(A, G0, G1, G2, G3) do {                                                    \
        A[0]=(__bf16)G0.x;  A[1]=(__bf16)G0.y;  A[2]=(__bf16)G0.z;  A[3]=(__bf16)G0.w;   \
        A[4]=(__bf16)G1.x;  A[5]=(__bf16)G1.y;  A[6]=(__bf16)G1.z;  A[7]=(__bf16)G1.w;   \
        A[8]=(__bf16)G2.x;  A[9]=(__bf16)G2.y;  A[10]=(__bf16)G2.z; A[11]=(__bf16)G2.w;  \
        A[12]=(__bf16)G3.x; A[13]=(__bf16)G3.y; A[14]=(__bf16)G3.z; A[15]=(__bf16)G3.w; } while (0)

#define LOAD_B(K0, B0, B1, B2, B3) do {                                          \
        B0 = load16bf(wbase + (size_t)( 0 + m) * DMODEL + (K0) + half * 16);     \
        B1 = load16bf(wbase + (size_t)(16 + m) * DMODEL + (K0) + half * 16);     \
        B2 = load16bf(wbase + (size_t)(32 + m) * DMODEL + (K0) + half * 16);     \
        B3 = load16bf(wbase + (size_t)(48 + m) * DMODEL + (K0) + half * 16); } while (0)

    v8f acc0 = (v8f){}, acc1 = (v8f){}, acc2 = (v8f){}, acc3 = (v8f){};

    float4 pf0, pf1, pf2, pf3, nf0, nf1, nf2, nf3;
    v16bf  pb0, pb1, pb2, pb3, nb0, nb1, nb2, nb3;

    LOAD_A(0, pf0, pf1, pf2, pf3);
    LOAD_B(0, pb0, pb1, pb2, pb3);

    for (int k0 = 0; k0 < DMODEL; k0 += 64) {
        const int k1 = k0 + 32;                                   // always valid
        const int k2 = (k0 + 64 < DMODEL) ? (k0 + 64) : k0;       // clamped prefetch

        // prefetch phase-1 tiles while phase-0 WMMAs run
        LOAD_A(k1, nf0, nf1, nf2, nf3);
        LOAD_B(k1, nb0, nb1, nb2, nb3);

        v16bf a0; CVT_A(a0, pf0, pf1, pf2, pf3);
        acc0 = wmma_bf16(a0, pb0, acc0);
        acc1 = wmma_bf16(a0, pb1, acc1);
        acc2 = wmma_bf16(a0, pb2, acc2);
        acc3 = wmma_bf16(a0, pb3, acc3);

        // prefetch phase-0 tiles of the next k-step
        LOAD_A(k2, pf0, pf1, pf2, pf3);
        LOAD_B(k2, pb0, pb1, pb2, pb3);

        v16bf a1; CVT_A(a1, nf0, nf1, nf2, nf3);
        acc0 = wmma_bf16(a1, nb0, acc0);
        acc1 = wmma_bf16(a1, nb1, acc1);
        acc2 = wmma_bf16(a1, nb2, acc2);
        acc3 = wmma_bf16(a1, nb3, acc3);
    }

    // Store: C/D layout -> element r is row (r + 8*half), column (n*16 + m).
    v8f accs[4] = { acc0, acc1, acc2, acc3 };
    if (mm == 2) {
        #pragma unroll
        for (int n = 0; n < 4; ++n)
            #pragma unroll
            for (int r = 0; r < 8; ++r) {
                const int row = row0 + r + 8 * half;
                const int h   = n * 16 + m;
                const int bb  = row >> 11;
                const int t   = row & (TSEQ - 1);
                vtb[((size_t)bb * HSIZE + h) * TSEQ + t] = (__bf16)accs[n][r];
            }
    } else {
        __bf16* dst = (mm == 0) ? qb : kb;
        #pragma unroll
        for (int n = 0; n < 4; ++n)
            #pragma unroll
            for (int r = 0; r < 8; ++r) {
                const int row = row0 + r + 8 * half;
                dst[(size_t)row * HSIZE + n * 16 + m] = (__bf16)accs[n][r];
            }
    }
#undef LOAD_A
#undef CVT_A
#undef LOAD_B
}

// ---------------------------------------------------------------------------
// Kernel 2: flash attention, one wave per (batch, 16-query tile).
//   K/V tiles software-prefetched one 32-key block ahead.
// ---------------------------------------------------------------------------
__global__ __launch_bounds__(32)
void attn_kernel(const __bf16* __restrict__ qb,
                 const __bf16* __restrict__ kb,
                 const __bf16* __restrict__ vtb,
                 float* __restrict__ out)
{
    __shared__ __bf16 pLds[16 * 32];   // P tile: 16 query rows x 32 keys

    const int lane = threadIdx.x & 31;
    const int m    = lane & 15;
    const int half = lane >> 4;
    const int qt   = blockIdx.x;       // query tile 0..127
    const int b    = blockIdx.y;       // batch 0..3
    const int q0   = qt * 16;

    const __bf16* q  = qb  + (size_t)b * TSEQ * HSIZE;
    const __bf16* k  = kb  + (size_t)b * TSEQ * HSIZE;
    const __bf16* vt = vtb + (size_t)b * HSIZE * TSEQ;

#define LOAD_K(KB, T00, T01, T10, T11) do {                                   \
        T00 = load16bf(k + (size_t)((KB) +  0 + m) * HSIZE +  0 + half * 16); \
        T01 = load16bf(k + (size_t)((KB) +  0 + m) * HSIZE + 32 + half * 16); \
        T10 = load16bf(k + (size_t)((KB) + 16 + m) * HSIZE +  0 + half * 16); \
        T11 = load16bf(k + (size_t)((KB) + 16 + m) * HSIZE + 32 + half * 16); } while (0)

#define LOAD_V(KB, T0, T1, T2, T3) do {                                       \
        T0 = load16bf(vt + (size_t)( 0 + m) * TSEQ + (KB) + half * 16);       \
        T1 = load16bf(vt + (size_t)(16 + m) * TSEQ + (KB) + half * 16);       \
        T2 = load16bf(vt + (size_t)(32 + m) * TSEQ + (KB) + half * 16);       \
        T3 = load16bf(vt + (size_t)(48 + m) * TSEQ + (KB) + half * 16); } while (0)

    // Preload Q tile (loop-invariant) in A layout: 2 chunks of K-dim 32.
    v16bf Qa0, Qa1;
    {
        const __bf16* qr = q + (size_t)(q0 + m) * HSIZE + half * 16;
        v8bf lo0 = *(const v8bf*)qr,        hi0 = *(const v8bf*)(qr + 8);
        v8bf lo1 = *(const v8bf*)(qr + 32), hi1 = *(const v8bf*)(qr + 40);
        Qa0 = __builtin_shufflevector(lo0, hi0, 0, 1, 2, 3, 4, 5, 6, 7,
                                                8, 9, 10, 11, 12, 13, 14, 15);
        Qa1 = __builtin_shufflevector(lo1, hi1, 0, 1, 2, 3, 4, 5, 6, 7,
                                                8, 9, 10, 11, 12, 13, 14, 15);
    }

    v8f o0 = (v8f){}, o1 = (v8f){}, o2 = (v8f){}, o3 = (v8f){};
    float mrow[8], lrow[8];
    #pragma unroll
    for (int r = 0; r < 8; ++r) { mrow[r] = -__builtin_inff(); lrow[r] = 0.0f; }

    const int nblocks = qt / 2 + 1;

    v16bf K00, K01, K10, K11, V0, V1, V2, V3;
    LOAD_K(0, K00, K01, K10, K11);
    LOAD_V(0, V0, V1, V2, V3);

    #pragma unroll 2
    for (int jb = 0; jb < nblocks; ++jb) {
        const int kb0   = jb * 32;
        const int knext = (jb + 1 < nblocks) ? kb0 + 32 : kb0;   // clamped

        // ---- S = Q * K^T (two 16-key subtiles, K-dim 64) ------------------
        v8f S0 = (v8f){}, S1 = (v8f){};
        S0 = wmma_bf16(Qa0, K00, S0);
        S0 = wmma_bf16(Qa1, K01, S0);
        S1 = wmma_bf16(Qa0, K10, S1);
        S1 = wmma_bf16(Qa1, K11, S1);

        // ---- prefetch next block's K and V tiles --------------------------
        v16bf nK00, nK01, nK10, nK11, nV0, nV1, nV2, nV3;
        LOAD_K(knext, nK00, nK01, nK10, nK11);
        LOAD_V(knext, nV0, nV1, nV2, nV3);

        // ---- scale (reference multiplies by sqrt(HS)=8) + causal mask -----
        #pragma unroll
        for (int r = 0; r < 8; ++r) {
            const int qrow = q0 + r + 8 * half;
            const float a = S0[r] * 8.0f;
            const float c = S1[r] * 8.0f;
            S0[r] = ((kb0 + m)      <= qrow) ? a : -__builtin_inff();
            S1[r] = ((kb0 + 16 + m) <= qrow) ? c : -__builtin_inff();
        }

        // ---- online softmax (row reductions across 16 lanes of the half) --
        float corr[8];
        #pragma unroll
        for (int r = 0; r < 8; ++r) {
            float v = fmaxf(S0[r], S1[r]);
            v = fmaxf(v, __shfl_xor(v, 1, 32));
            v = fmaxf(v, __shfl_xor(v, 2, 32));
            v = fmaxf(v, __shfl_xor(v, 4, 32));
            v = fmaxf(v, __shfl_xor(v, 8, 32));
            const float mnew = fmaxf(mrow[r], v);
            corr[r] = __expf(mrow[r] - mnew);
            mrow[r] = mnew;
        }
        #pragma unroll
        for (int r = 0; r < 8; ++r) {
            const float p0 = __expf(S0[r] - mrow[r]);
            const float p1 = __expf(S1[r] - mrow[r]);
            S0[r] = p0; S1[r] = p1;
            float v = p0 + p1;
            v += __shfl_xor(v, 1, 32);
            v += __shfl_xor(v, 2, 32);
            v += __shfl_xor(v, 4, 32);
            v += __shfl_xor(v, 8, 32);
            lrow[r] = lrow[r] * corr[r] + v;
        }
        #pragma unroll
        for (int r = 0; r < 8; ++r) {
            o0[r] *= corr[r]; o1[r] *= corr[r];
            o2[r] *= corr[r]; o3[r] *= corr[r];
        }

        // ---- P: C/D layout -> A layout via LDS bounce (bf16) --------------
        #pragma unroll
        for (int r = 0; r < 8; ++r) {
            pLds[(r + 8 * half) * 32 +  0 + m] = (__bf16)S0[r];
            pLds[(r + 8 * half) * 32 + 16 + m] = (__bf16)S1[r];
        }
        __syncthreads();   // single-wave WG: cheap barrier

        v16bf Pa;
        {
            const __bf16* pr = &pLds[m * 32 + half * 16];
            v8bf lo = *(const v8bf*)pr;
            v8bf hi = *(const v8bf*)(pr + 8);
            Pa = __builtin_shufflevector(lo, hi, 0, 1, 2, 3, 4, 5, 6, 7,
                                                 8, 9, 10, 11, 12, 13, 14, 15);
        }
        __syncthreads();

        // ---- O += P * V ---------------------------------------------------
        o0 = wmma_bf16(Pa, V0, o0);
        o1 = wmma_bf16(Pa, V1, o1);
        o2 = wmma_bf16(Pa, V2, o2);
        o3 = wmma_bf16(Pa, V3, o3);

        // ---- rotate prefetched tiles (SSA copies, eliminated by unroll) ---
        K00 = nK00; K01 = nK01; K10 = nK10; K11 = nK11;
        V0 = nV0; V1 = nV1; V2 = nV2; V3 = nV3;
    }

    // ---- finalize: divide by l, store f32 ---------------------------------
    v8f os[4] = { o0, o1, o2, o3 };
    #pragma unroll
    for (int r = 0; r < 8; ++r) {
        const float inv = 1.0f / lrow[r];
        const int row = q0 + r + 8 * half;
        #pragma unroll
        for (int n = 0; n < 4; ++n)
            out[((size_t)b * TSEQ + row) * HSIZE + n * 16 + m] = os[n][r] * inv;
    }
#undef LOAD_K
#undef LOAD_V
}

// ---------------------------------------------------------------------------
extern "C" void kernel_launch(void* const* d_in, const int* in_sizes, int n_in,
                              void* d_out, int out_size, void* d_ws, size_t ws_size,
                              hipStream_t stream) {
    const float* x  = (const float*)d_in[0];
    const float* Wk = (const float*)d_in[1];
    const float* Wq = (const float*)d_in[2];
    const float* Wv = (const float*)d_in[3];
    float* out = (float*)d_out;

    const size_t elems = (size_t)BATCH * TSEQ * HSIZE;
    __bf16* qbuf = (__bf16*)d_ws;
    __bf16* kbuf = qbuf + elems;
    __bf16* vtb  = kbuf + elems;
    __bf16* wt   = vtb + elems;    // [3][HS][D]

    wtrans_kernel<<<dim3(16, 3), dim3(256), 0, stream>>>(Wq, Wk, Wv, wt);

    proj_kernel<<<dim3((BATCH * TSEQ) / 16, 3), dim3(32), 0, stream>>>(
        x, wt, qbuf, kbuf, vtb);

    attn_kernel<<<dim3(TSEQ / 16, BATCH), dim3(32), 0, stream>>>(
        qbuf, kbuf, vtb, out);
}
